// VLMSDeformAttnLayer_49392123904324
// MI455X (gfx1250) — matrix operand order repaired and checked by reading, not compiled, not run on hardware
//
#include <hip/hip_runtime.h>
#include <hip/hip_bf16.h>
#include <math.h>

// ---------------------------------------------------------------------------
// VLMSDeformAttnLayer for gfx1250 (MI455X).
// GEMMs: bf16 WMMA (v_wmma_f32_16x16x32_bf16), f32 accumulation.
//   Block tile 128x128, K-step 32; 8 waves (2x4); wave tile 64x32 -> 8 WMMA/iter.
// Softmax / LayerNorm / deformable bilinear sampling: f32 VALU.
// ---------------------------------------------------------------------------

typedef __bf16 bf16;
typedef __attribute__((ext_vector_type(16))) __bf16 v16bf;
typedef __attribute__((ext_vector_type(8)))  __bf16 v8bf;
typedef __attribute__((ext_vector_type(8)))  float  v8f;

static constexpr int kB     = 4;
static constexpr int kSLang = 32;
static constexpr int kE     = 256;
static constexpr int kHeads = 8;
static constexpr int kHdA   = 128;    // ATTN_DIM / HEADS
static constexpr int kADim  = 1024;
static constexpr int kMDim  = 1024;
static constexpr int kLVis  = 11253;  // 92^2 + 46^2 + 23^2 + 12^2
static constexpr int kNV    = kB * kLVis;   // 45012
static constexpr int kNL    = kB * kSLang;  // 128

// ------------------------------- GEMM --------------------------------------
#define GTM 128
#define GTN 128
#define GTK 32
#define BPAD 8

__global__ __launch_bounds__(256)
void gemm_bf16_wmma(const bf16* __restrict__ A, const bf16* __restrict__ W,
                    const float* __restrict__ bias,
                    float* __restrict__ C, bf16* __restrict__ Cbf,
                    int M, int N, int K, float alpha, int relu)
{
    __shared__ bf16 sA[GTM][GTK];
    __shared__ bf16 sB[GTN][GTK + BPAD];   // B stored K-transposed (N rows x K cols)

    const int tid  = threadIdx.x;
    const int lane = tid & 31;
    const int wave = tid >> 5;
    const int wm   = wave & 1;     // 2 waves along M (64 rows each)
    const int wn   = wave >> 1;    // 4 waves along N (32 cols each)
    const int rowBase = blockIdx.y * GTM;
    const int colBase = blockIdx.x * GTN;

    v8f acc[4][2] = {};

    // B staging map: thread -> (k-pair, 8 columns)
    const int kp = tid >> 4;           // 0..15 -> k rows {2kp, 2kp+1}
    const int bn0 = (tid & 15) * 8;    // 0..120

    for (int k0 = 0; k0 < K; k0 += GTK) {
        __syncthreads();
        // --- stage A tile (row-major), 2 chunks of 8 bf16 per thread ---
        #pragma unroll
        for (int i = 0; i < 2; ++i) {
            int chunk = tid + i * 256;         // 512 chunks cover 128x32
            int r = chunk >> 2;                // 0..127
            int c = (chunk & 3) * 8;           // 0,8,16,24
            int gr = rowBase + r;
            if (gr < M) {
                v8bf t = *(const v8bf*)(A + (size_t)gr * K + k0 + c);
                *(v8bf*)&sA[r][c] = t;
                if (k0 + GTK < K)
                    __builtin_prefetch(A + (size_t)gr * K + k0 + GTK + c, 0, 0);
            } else {
                v8bf z = {};
                *(v8bf*)&sA[r][c] = z;
            }
        }
        // --- stage B tile transposed: two K-rows packed into dword stores ---
        {
            const bf16* wp = W + (size_t)(k0 + 2 * kp) * N + colBase + bn0;
            v8bf r0 = *(const v8bf*)wp;
            v8bf r1 = *(const v8bf*)(wp + N);
            #pragma unroll
            for (int j = 0; j < 8; ++j) {
                union { unsigned u; __bf16 h[2]; } pk;
                pk.h[0] = r0[j];
                pk.h[1] = r1[j];
                *(unsigned*)&sB[bn0 + j][2 * kp] = pk.u;   // 4B-aligned
            }
            if (k0 + GTK < K)
                __builtin_prefetch(W + (size_t)(k0 + GTK + 2 * kp) * N + colBase + bn0, 0, 0);
        }
        __syncthreads();

        // --- fragments (wave32 WMMA layouts per CDNA5 ISA 7.12.2) ---
        const int  lr = lane & 15;
        const bool hi = lane >= 16;
        v16bf afrag[4], bfrag[2];
        #pragma unroll
        for (int i = 0; i < 4; ++i) {
            // A 16x32: lane<16 holds K{0..7,16..23}; lane>=16 holds K{8..15,24..31}
            const bf16* ap = &sA[wm * 64 + i * 16 + lr][0];
            v8bf lo = *(const v8bf*)(ap + (hi ? 8 : 0));
            v8bf hv = *(const v8bf*)(ap + (hi ? 8 : 0) + 16);
            v16bf a;
            #pragma unroll
            for (int j = 0; j < 8; ++j) { a[j] = lo[j]; a[8 + j] = hv[j]; }
            afrag[i] = a;
        }
        #pragma unroll
        for (int i = 0; i < 2; ++i) {
            // B 32x16: lane<16 holds K{0..7,16..23} of column lane; lane>=16 holds K{8..15,24..31}
            const bf16* bp = &sB[wn * 32 + i * 16 + lr][hi ? 8 : 0];
            v8bf b0 = *(const v8bf*)bp;
            v8bf b1 = *(const v8bf*)(bp + 16);
            v16bf bb;
            #pragma unroll
            for (int j = 0; j < 8; ++j) { bb[j] = b0[j]; bb[8 + j] = b1[j]; }
            bfrag[i] = bb;
        }
        #pragma unroll
        for (int mi = 0; mi < 4; ++mi)
            #pragma unroll
            for (int ni = 0; ni < 2; ++ni)
                acc[mi][ni] = __builtin_amdgcn_wmma_f32_16x16x32_bf16(
                    false, afrag[mi], false, bfrag[ni],
                    (short)0, acc[mi][ni], false, false);
    }

    // --- epilogue: C/D layout VGPR j -> M=j(+8 for hi lanes), N = lane&15 ---
    const int  lr = lane & 15;
    const bool hi = lane >= 16;
    #pragma unroll
    for (int mi = 0; mi < 4; ++mi) {
        #pragma unroll
        for (int ni = 0; ni < 2; ++ni) {
            int gc = colBase + wn * 32 + ni * 16 + lr;
            float bia = bias ? bias[gc] : 0.0f;
            #pragma unroll
            for (int j = 0; j < 8; ++j) {
                int gr = rowBase + wm * 64 + mi * 16 + j + (hi ? 8 : 0);
                if (gr < M) {
                    float v = alpha * (acc[mi][ni][j] + bia);
                    if (relu) v = fmaxf(v, 0.0f);
                    if (C)   C[(size_t)gr * N + gc] = v;
                    if (Cbf) Cbf[(size_t)gr * N + gc] = (bf16)v;
                }
            }
        }
    }
}

// ----------------------- attention (scores+softmax+AV) ---------------------
// One block per (q, h, b). Scores row lives entirely in LDS (Lk <= 11264).
#define MAX_LK 11264
__global__ __launch_bounds__(128)
void attn_softmax_av(const float* __restrict__ Q, const float* __restrict__ Kb,
                     const float* __restrict__ Vb, const int* __restrict__ kmask,
                     float* __restrict__ O, int Lq, int Lk)
{
    __shared__ float qrow[kHdA];
    __shared__ float sc[MAX_LK];
    __shared__ float red[4];

    const int q = blockIdx.x, h = blockIdx.y, b = blockIdx.z;
    const int tid = threadIdx.x, lane = tid & 31, wave = tid >> 5;
    const size_t qoff = ((size_t)b * Lq + q) * kADim + (size_t)h * kHdA;

    qrow[tid] = Q[qoff + tid];
    __syncthreads();

    float lmax = -3.0e38f;
    for (int k = tid; k < Lk; k += 128) {
        const float4* kr = (const float4*)(Kb + ((size_t)b * Lk + k) * kADim + (size_t)h * kHdA);
        const float4* qr = (const float4*)qrow;
        float s = 0.f;
        #pragma unroll 8
        for (int d = 0; d < kHdA / 4; ++d) {
            float4 kv = kr[d], qv = qr[d];
            s += qv.x * kv.x + qv.y * kv.y + qv.z * kv.z + qv.w * kv.w;
        }
        s = fminf(fmaxf(s, -50000.f), 50000.f);
        if (kmask[(size_t)b * Lk + k] == 0) s = -1.0e9f;
        sc[k] = s;
        lmax = fmaxf(lmax, s);
    }
    #pragma unroll
    for (int o = 16; o > 0; o >>= 1) lmax = fmaxf(lmax, __shfl_xor(lmax, o, 32));
    if (lane == 0) red[wave] = lmax;
    __syncthreads();
    lmax = fmaxf(fmaxf(red[0], red[1]), fmaxf(red[2], red[3]));

    float lsum = 0.f;
    for (int k = tid; k < Lk; k += 128) {
        float e = __expf(sc[k] - lmax);
        sc[k] = e;
        lsum += e;
    }
    #pragma unroll
    for (int o = 16; o > 0; o >>= 1) lsum += __shfl_xor(lsum, o, 32);
    __syncthreads();
    if (lane == 0) red[wave] = lsum;
    __syncthreads();
    lsum = red[0] + red[1] + red[2] + red[3];
    const float inv = (lsum > 0.f) ? 1.0f / lsum : 0.f;

    const int d = tid;
    float acc = 0.f;
    for (int k = 0; k < Lk; ++k)
        acc += sc[k] * Vb[((size_t)b * Lk + k) * kADim + (size_t)h * kHdA + d];
    O[qoff + d] = acc * inv;
}

// ------------------------------ LayerNorm ----------------------------------
__global__ __launch_bounds__(256)
void ln_res(const float* __restrict__ X, const float* __restrict__ R,
            const float* __restrict__ g, const float* __restrict__ bb,
            float* __restrict__ outF, bf16* __restrict__ outB)
{
    __shared__ float red[8];
    const int row = blockIdx.x, tid = threadIdx.x, lane = tid & 31, wave = tid >> 5;
    const size_t off = (size_t)row * kE + tid;
    float v = X[off] + (R ? R[off] : 0.f);

    float s = v;
    #pragma unroll
    for (int o = 16; o > 0; o >>= 1) s += __shfl_xor(s, o, 32);
    if (lane == 0) red[wave] = s;
    __syncthreads();
    float mean = 0.f;
    #pragma unroll
    for (int i = 0; i < 8; ++i) mean += red[i];
    mean *= (1.0f / kE);

    float dv = v - mean;
    float s2 = dv * dv;
    #pragma unroll
    for (int o = 16; o > 0; o >>= 1) s2 += __shfl_xor(s2, o, 32);
    __syncthreads();
    if (lane == 0) red[wave] = s2;
    __syncthreads();
    float var = 0.f;
    #pragma unroll
    for (int i = 0; i < 8; ++i) var += red[i];
    var *= (1.0f / kE);

    float y = dv * rsqrtf(var + 1e-5f) * g[tid] + bb[tid];
    if (outF) outF[off] = y;
    if (outB) outB[off] = (bf16)y;
}

// --------------------------- small utilities -------------------------------
__global__ void conv_f32_bf16(const float* __restrict__ x, bf16* __restrict__ y, long long n)
{
    long long i = (long long)blockIdx.x * blockDim.x + threadIdx.x;
    long long st = (long long)gridDim.x * blockDim.x;
    for (; i < n; i += st) y[i] = (bf16)x[i];
}

__global__ void add_f32_bf16(const float* __restrict__ a, const float* __restrict__ b,
                             bf16* __restrict__ y, long long n)
{
    long long i = (long long)blockIdx.x * blockDim.x + threadIdx.x;
    long long st = (long long)gridDim.x * blockDim.x;
    for (; i < n; i += st) y[i] = (bf16)(a[i] + b[i]);
}

__global__ void inv_mask_kernel(const int* __restrict__ m, int* __restrict__ o, int n)
{
    int i = blockIdx.x * blockDim.x + threadIdx.x;
    if (i < n) o[i] = (m[i] == 0) ? 1 : 0;
}

__global__ void mask_rows(float* __restrict__ v, const int* __restrict__ pad)
{
    const int r = blockIdx.x;
    const float m = pad[r] ? 0.f : 1.f;
    v[(size_t)r * kE + threadIdx.x] *= m;
}

// ------------------------ deformable bilinear sampling ---------------------
// One block per (b*L+q); 256 threads = 8 heads x 32 channels.
__global__ __launch_bounds__(256)
void deform_sample(const float* __restrict__ off, const float* __restrict__ awl,
                   const float* __restrict__ val, const float* __restrict__ ref,
                   float* __restrict__ out)
{
    const int Hs[4] = {92, 46, 23, 12};
    const int starts[4] = {0, 8464, 10580, 11109};

    const size_t row = blockIdx.x;                // b*L_VIS + q
    const int tid = threadIdx.x;
    const int h = tid >> 5, d = tid & 31;
    const int b = (int)(row / kLVis);

    // softmax over the 16 (level, point) logits for this head
    float lg[16];
    float m = -3.0e38f;
    #pragma unroll
    for (int j = 0; j < 16; ++j) { lg[j] = awl[row * 128 + h * 16 + j]; m = fmaxf(m, lg[j]); }
    float ssum = 0.f;
    #pragma unroll
    for (int j = 0; j < 16; ++j) { lg[j] = __expf(lg[j] - m); ssum += lg[j]; }
    const float invs = 1.0f / ssum;

    float acc = 0.f;
    #pragma unroll
    for (int l = 0; l < 4; ++l) {
        const int Hl = Hs[l], Wl = Hs[l], st = starts[l];
        const float rx = ref[(row * 4 + l) * 2 + 0];
        const float ry = ref[(row * 4 + l) * 2 + 1];
        #pragma unroll
        for (int p = 0; p < 4; ++p) {
            const int oi = (((h * 4) + l) * 4 + p) * 2;
            const float ox = off[row * 256 + oi + 0];
            const float oy = off[row * 256 + oi + 1];
            const float x = (rx + ox / (float)Wl) * Wl - 0.5f;
            const float y = (ry + oy / (float)Hl) * Hl - 0.5f;
            const float x0f = floorf(x), y0f = floorf(y);
            const int x0 = (int)x0f, y0 = (int)y0f, x1 = x0 + 1, y1 = y0 + 1;
            const float wx1 = x - x0f, wx0 = 1.f - wx1;
            const float wy1 = y - y0f, wy0 = 1.f - wy1;
            float g = 0.f;
            if (x0 >= 0 && x0 < Wl && y0 >= 0 && y0 < Hl)
                g += wx0 * wy0 * val[((size_t)b * kLVis + st + y0 * Wl + x0) * kE + h * 32 + d];
            if (x1 >= 0 && x1 < Wl && y0 >= 0 && y0 < Hl)
                g += wx1 * wy0 * val[((size_t)b * kLVis + st + y0 * Wl + x1) * kE + h * 32 + d];
            if (x0 >= 0 && x0 < Wl && y1 >= 0 && y1 < Hl)
                g += wx0 * wy1 * val[((size_t)b * kLVis + st + y1 * Wl + x0) * kE + h * 32 + d];
            if (x1 >= 0 && x1 < Wl && y1 >= 0 && y1 < Hl)
                g += wx1 * wy1 * val[((size_t)b * kLVis + st + y1 * Wl + x1) * kE + h * 32 + d];
            acc += (lg[l * 4 + p] * invs) * g;
        }
    }
    out[row * 256 + h * 32 + d] = acc;
}

// ------------------------------- host side ---------------------------------
static inline void launch_gemm(hipStream_t s, const bf16* A, const bf16* W,
                               const float* bias, float* C, bf16* Cb,
                               int M, int N, int K, float alpha, int relu)
{
    dim3 g(N / GTN, (M + GTM - 1) / GTM);
    gemm_bf16_wmma<<<g, 256, 0, s>>>(A, W, bias, C, Cb, M, N, K, alpha, relu);
}

static inline void launch_conv(hipStream_t s, const float* x, bf16* y, long long n)
{
    conv_f32_bf16<<<2048, 256, 0, s>>>(x, y, n);
}

extern "C" void kernel_launch(void* const* d_in, const int* in_sizes, int n_in,
                              void* d_out, int out_size, void* d_ws, size_t ws_size,
                              hipStream_t stream)
{
    (void)in_sizes; (void)n_in; (void)out_size; (void)ws_size;

    // inputs (setup_inputs() insertion order; params flattened depth-first)
    const float* in_vis   = (const float*)d_in[0];
    const float* in_lang  = (const float*)d_in[1];
    const int*   in_lmask = (const int*)  d_in[2];
    const float* in_vpos  = (const float*)d_in[3];
    const int*   in_vmask = (const int*)  d_in[4];
    const float* in_ref   = (const float*)d_in[5];
    auto F = [&](int i) { return (const float*)d_in[i]; };

    // workspace carving
    char* wsb = (char*)d_ws;
    size_t off = 0;
    auto carve = [&](size_t bytes) -> char* {
        char* p = wsb + off;
        off += (bytes + 255) & ~(size_t)255;
        return p;
    };
    const size_t NVE = (size_t)kNV * kE, NVA = (size_t)kNV * kADim;
    const size_t NLE = (size_t)kNL * kE, NLA = (size_t)kNL * kADim;

    float* F1   = (float*)carve(NVA * 4);   // K_v2l -> Q_l2v -> {DVAL, DSAMP}
    float* F2   = (float*)carve(NVA * 4);   // V_v2l -> attnVis -> TMPA
    bf16*  HIDB = (bf16*) carve(NVA * 2);
    bf16*  VPB  = (bf16*) carve(NVE * 2);
    bf16*  VISB = (bf16*) carve(NVE * 2);
    float* VIS2 = (float*)carve(NVE * 4);
    float* DOFF = (float*)carve(NVE * 4);   // aliased as VIS3 after sampling
    float* DAW  = (float*)carve((size_t)kNV * 128 * 4);
    int*   INVM = (int*)  carve((size_t)kNV * 4);
    float* LQ   = (float*)carve(NLA * 4);
    float* LK   = (float*)carve(NLA * 4);
    float* LVv  = (float*)carve(NLA * 4);
    float* LO   = (float*)carve(NLA * 4);
    float* LT   = (float*)carve(NLE * 4);
    float* LC   = (float*)carve(NLE * 4);
    bf16*  LBF  = (bf16*) carve(NLA * 2);
    bf16*  LHBF = (bf16*) carve(NLA * 2);

    float* DVAL  = F1;
    float* DSAMP = F1 + NVE;
    float* TMPA  = F2;
    float* VIS3  = DOFF;

    // bf16 weights
    struct WConv { int idx; bf16* dst; size_t n; };
    bf16* w_v2l_q = (bf16*)carve(256 * 1024 * 2);
    bf16* w_v2l_k = (bf16*)carve(256 * 1024 * 2);
    bf16* w_v2l_v = (bf16*)carve(256 * 1024 * 2);
    bf16* w_v2l_o = (bf16*)carve(1024 * 256 * 2);
    bf16* w_ls_q  = (bf16*)carve(256 * 1024 * 2);
    bf16* w_ls_k  = (bf16*)carve(256 * 1024 * 2);
    bf16* w_ls_v  = (bf16*)carve(256 * 1024 * 2);
    bf16* w_ls_o  = (bf16*)carve(1024 * 256 * 2);
    bf16* w_l2v_q = (bf16*)carve(256 * 1024 * 2);
    bf16* w_l2v_k = (bf16*)carve(256 * 1024 * 2);
    bf16* w_l2v_v = (bf16*)carve(256 * 1024 * 2);
    bf16* w_l2v_o = (bf16*)carve(1024 * 256 * 2);
    bf16* w_off   = (bf16*)carve(256 * 256 * 2);
    bf16* w_attn  = (bf16*)carve(256 * 128 * 2);
    bf16* w_val   = (bf16*)carve(256 * 256 * 2);
    bf16* w_outp  = (bf16*)carve(256 * 256 * 2);
    bf16* w_vl1   = (bf16*)carve(256 * 1024 * 2);
    bf16* w_vl2   = (bf16*)carve(1024 * 256 * 2);
    bf16* w_ll1   = (bf16*)carve(256 * 1024 * 2);
    bf16* w_ll2   = (bf16*)carve(1024 * 256 * 2);

    const WConv wc[] = {
        {6,  w_v2l_q, 256*1024}, {8,  w_v2l_k, 256*1024}, {10, w_v2l_v, 256*1024}, {12, w_v2l_o, 1024*256},
        {14, w_ls_q,  256*1024}, {16, w_ls_k,  256*1024}, {18, w_ls_v,  256*1024}, {20, w_ls_o,  1024*256},
        {22, w_l2v_q, 256*1024}, {24, w_l2v_k, 256*1024}, {26, w_l2v_v, 256*1024}, {28, w_l2v_o, 1024*256},
        {30, w_off,   256*256},  {32, w_attn,  256*128},  {34, w_val,   256*256},  {36, w_outp,  256*256},
        {38, w_vl1,   256*1024}, {40, w_vl2,   1024*256}, {42, w_ll1,   256*1024}, {44, w_ll2,   1024*256},
    };
    for (const auto& w : wc) launch_conv(stream, F(w.idx), w.dst, (long long)w.n);

    float* outVis  = (float*)d_out;
    float* outLang = outVis + NVE;
    const float SCALE = 0.08838834764831845f;   // 128^-0.5

    inv_mask_kernel<<<(kNV + 255) / 256, 256, 0, stream>>>(in_vmask, INVM, kNV);
    add_f32_bf16<<<2048, 256, 0, stream>>>(in_vis, in_vpos, VPB, (long long)NVE);
    launch_conv(stream, in_vis,  VISB, (long long)NVE);
    launch_conv(stream, in_lang, LBF,  (long long)NLE);

    // ---- v2l cross attention (lang queries vis) ----
    launch_gemm(stream, LBF,  w_v2l_q, F(7),  LQ, nullptr, kNL, kADim, kE, SCALE, 0);
    launch_gemm(stream, VPB,  w_v2l_k, F(9),  F1, nullptr, kNV, kADim, kE, 1.f, 0);
    launch_gemm(stream, VISB, w_v2l_v, F(11), F2, nullptr, kNV, kADim, kE, 1.f, 0);
    attn_softmax_av<<<dim3(kSLang, kHeads, kB), 128, 0, stream>>>(LQ, F1, F2, INVM, LO, kSLang, kLVis);
    launch_conv(stream, LO, LBF, (long long)NLA);
    launch_gemm(stream, LBF, w_v2l_o, F(13), LT, nullptr, kNL, kE, kADim, 1.f, 0);
    ln_res<<<kNL, 256, 0, stream>>>(LT, in_lang, F(52), F(53), LC, LBF);      // ln.l1

    // ---- lang self attention ----
    launch_gemm(stream, LBF, w_ls_q, F(15), LQ,  nullptr, kNL, kADim, kE, SCALE, 0);
    launch_gemm(stream, LBF, w_ls_k, F(17), LK,  nullptr, kNL, kADim, kE, 1.f, 0);
    launch_gemm(stream, LBF, w_ls_v, F(19), LVv, nullptr, kNL, kADim, kE, 1.f, 0);
    attn_softmax_av<<<dim3(kSLang, kHeads, kB), 128, 0, stream>>>(LQ, LK, LVv, in_lmask, LO, kSLang, kSLang);
    launch_conv(stream, LO, LHBF, (long long)NLA);
    launch_gemm(stream, LHBF, w_ls_o, F(21), LT, nullptr, kNL, kE, kADim, 1.f, 0);
    ln_res<<<kNL, 256, 0, stream>>>(LT, LC, F(54), F(55), LC, LBF);           // ln.l2

    // ---- lang MLP ----
    launch_gemm(stream, LBF,  w_ll1, F(43), nullptr, LHBF, kNL, kMDim, kE, 1.f, 1);
    launch_gemm(stream, LHBF, w_ll2, F(45), LT, nullptr, kNL, kE, kMDim, 1.f, 0);
    ln_res<<<kNL, 256, 0, stream>>>(LT, LC, F(56), F(57), outLang, LBF);      // ln.l3 -> final lang

    // ---- l2v cross attention (vis queries lang) ----
    launch_gemm(stream, VPB, w_l2v_q, F(23), F1,  nullptr, kNV, kADim, kE, SCALE, 0);
    launch_gemm(stream, LBF, w_l2v_k, F(25), LK,  nullptr, kNL, kADim, kE, 1.f, 0);
    launch_gemm(stream, LBF, w_l2v_v, F(27), LVv, nullptr, kNL, kADim, kE, 1.f, 0);
    attn_softmax_av<<<dim3(kLVis, kHeads, kB), 128, 0, stream>>>(F1, LK, LVv, in_lmask, F2, kLVis, kSLang);
    launch_conv(stream, F2, HIDB, (long long)NVA);
    launch_gemm(stream, HIDB, w_l2v_o, F(29), TMPA, nullptr, kNV, kE, kADim, 1.f, 0);
    ln_res<<<kNV, 256, 0, stream>>>(TMPA, in_vis, F(48), F(49), VIS2, VISB);  // ln.v2

    // ---- multi-scale deformable attention ----
    add_f32_bf16<<<2048, 256, 0, stream>>>(VIS2, in_vpos, VPB, (long long)NVE);
    launch_gemm(stream, VISB, w_val,  F(35), DVAL, nullptr, kNV, kE, kE, 1.f, 0);
    mask_rows<<<kNV, 256, 0, stream>>>(DVAL, in_vmask);
    launch_gemm(stream, VPB, w_off,  F(31), DOFF, nullptr, kNV, kE,  kE, 1.f, 0);
    launch_gemm(stream, VPB, w_attn, F(33), DAW,  nullptr, kNV, 128, kE, 1.f, 0);
    deform_sample<<<kNV, 256, 0, stream>>>(DOFF, DAW, DVAL, in_ref, DSAMP);
    launch_conv(stream, DSAMP, VISB, (long long)NVE);
    launch_gemm(stream, VISB, w_outp, F(37), TMPA, nullptr, kNV, kE, kE, 1.f, 0);
    ln_res<<<kNV, 256, 0, stream>>>(TMPA, VIS2, F(46), F(47), VIS3, VISB);    // ln.v1

    // ---- vis MLP ----
    launch_gemm(stream, VISB, w_vl1, F(39), nullptr, HIDB, kNV, kMDim, kE, 1.f, 1);
    launch_gemm(stream, HIDB, w_vl2, F(41), TMPA, nullptr, kNV, kE, kMDim, 1.f, 0);
    ln_res<<<kNV, 256, 0, stream>>>(TMPA, VIS3, F(50), F(51), outVis, nullptr); // ln.v3 -> final vis
}
